// ALIGNN_PyG_10831907521231
// MI455X (gfx1250) — matrix operand order repaired and checked by reading, not compiled, not run on hardware
//
#include <hip/hip_runtime.h>

// ---------------------------------------------------------------------------
// ALIGNN forward for MI455X (gfx1250): all 256x256 GEMMs on the bf16 WMMA pipe
// (v_wmma_f32_16x16x32_bf16), irregular graph ops as atomic scatter kernels.
// Node-side GEMMs (4 weights x same input) fused: A tile staged in LDS once.
// ---------------------------------------------------------------------------

#define H      256
#define NNODE  10000
#define NEDGE  120000
#define NANG   360000
#define NB     64

typedef __attribute__((ext_vector_type(16))) __bf16 v16bf;
typedef __attribute__((ext_vector_type(8)))  __bf16 v8bf;
typedef __attribute__((ext_vector_type(8)))  float  v8f;

__device__ __forceinline__ float silu_f(float v) { return v / (1.0f + __expf(-v)); }

// ------------------------------ elementwise --------------------------------

__global__ void zero_f32(float* p, long n) {
    long i = (long)blockIdx.x * blockDim.x + threadIdx.x;
    long s = (long)gridDim.x * blockDim.x;
    for (; i < n; i += s) p[i] = 0.0f;
}

__global__ void cast_bf16(const float* __restrict__ in, __bf16* __restrict__ out, long n) {
    long i = (long)blockIdx.x * blockDim.x + threadIdx.x;
    long s = (long)gridDim.x * blockDim.x;
    for (; i < n; i += s) out[i] = (__bf16)in[i];
}

__global__ void add_inplace(float* __restrict__ dst, const float* __restrict__ a, long n) {
    long i = (long)blockIdx.x * blockDim.x + threadIdx.x;
    long s = (long)gridDim.x * blockDim.x;
    for (; i < n; i += s) dst[i] += a[i];
}

// ------------------- weight packing: WMMA-fragment-major -------------------
// B fragment for v_wmma_f32_16x16x32_bf16 in this code's k order (A matches): lane l holds column n=l&15,
// element i -> K = ktile*32 + 16*(l>=16) + i.  Pack so each lane's 16 bf16
// are one contiguous 32B chunk.
__global__ void pack_w(const float* __restrict__ W, __bf16* __restrict__ Wp, long total) {
    long idx = (long)blockIdx.x * blockDim.x + threadIdx.x;
    long s = (long)gridDim.x * blockDim.x;
    for (; idx < total; idx += s) {
        long mat = idx >> 16;          // 65536 elems per 256x256 matrix
        int  rem = (int)(idx & 65535);
        int  k = rem >> 8, n = rem & 255;
        int  kt = k >> 5, kr = k & 31;
        int  hi = kr >> 4, i = kr & 15;
        long p = mat * 65536 + (((long)(kt * H + n) * 2 + hi) * 16 + i);
        Wp[p] = (__bf16)W[idx];
    }
}

// --------------------------- WMMA GEMM core --------------------------------
// One wave computes a 16(M) x 64(N) tile: 4 accumulators, 8 K-steps.
__device__ __forceinline__ void wmma_strip(const __bf16 (*As)[264],
                                           const __bf16* __restrict__ Wp,
                                           const float* __restrict__ bias,
                                           float* __restrict__ C,
                                           int rowBase, int nrows,
                                           int wave, int lane) {
    const int hi = lane >> 4;
    const int m  = lane & 15;
    v8f acc[4];
    for (int t = 0; t < 4; t++)
        for (int r = 0; r < 8; r++) acc[t][r] = 0.0f;

    for (int kk = 0; kk < H; kk += 32) {
        // A fragment: lane(0-15) K in {kk..kk+15};
        //             lane(16-31) K in {kk+16..kk+31}
        union { v16bf v; v8bf h[2]; } au;
        au.h[0] = *(const v8bf*)&As[m][kk + 16 * hi];
        au.h[1] = *(const v8bf*)&As[m][kk + 16 * hi + 8];
        const int kt = kk >> 5;
#pragma unroll
        for (int t = 0; t < 4; t++) {
            int n = wave * 64 + t * 16 + m;
            v16bf bv = *(const v16bf*)(Wp + ((size_t)(kt * H + n) * 2 + hi) * 16);
            acc[t] = __builtin_amdgcn_wmma_f32_16x16x32_bf16(
                false, au.v, false, bv, (short)0, acc[t], false, false);
        }
    }

#pragma unroll
    for (int t = 0; t < 4; t++) {
        int n = wave * 64 + t * 16 + m;
        float bv = bias[n];
#pragma unroll
        for (int r = 0; r < 8; r++) {
            int row = rowBase + r + 8 * hi;
            if (row < nrows) C[(size_t)row * H + n] = acc[t][r] + bv;
        }
    }
}

__device__ __forceinline__ void stage_a_tile(__bf16 (*As)[264],
                                             const __bf16* __restrict__ A,
                                             int rowBase, int nrows, int tid) {
    for (int c = tid; c < 16 * 32; c += 128) {
        int r = c >> 5, cc = (c & 31) * 8;
        int row = rowBase + r;
        v8bf v;
        if (row < nrows) {
            v = *(const v8bf*)(A + (size_t)row * H + cc);
        } else {
            for (int i = 0; i < 8; i++) v[i] = (__bf16)0.0f;
        }
        *(v8bf*)&As[r][cc] = v;
    }
}

// single-matrix GEMM (used for the big edge-gate GEMM)
__global__ void gemm_wmma(const __bf16* __restrict__ A, const __bf16* __restrict__ Wp,
                          const float* __restrict__ bias, float* __restrict__ C, int nrows) {
    __shared__ __align__(16) __bf16 As[16][264];
    const int tid = threadIdx.x;
    const int rowBase = blockIdx.x * 16;
    stage_a_tile(As, A, rowBase, nrows, tid);
    __syncthreads();
    wmma_strip(As, Wp, bias, C, rowBase, nrows, tid >> 5, tid & 31);
}

// fused: 4 weight matrices (indices 0,1,4,3 of the layer) over one A tile.
__global__ void gemm_wmma4(const __bf16* __restrict__ A, const __bf16* __restrict__ WpL,
                           const float* __restrict__ bL,
                           float* __restrict__ C0, float* __restrict__ C1,
                           float* __restrict__ C2, float* __restrict__ C3, int nrows) {
    __shared__ __align__(16) __bf16 As[16][264];
    const int tid = threadIdx.x;
    const int rowBase = blockIdx.x * 16;
    stage_a_tile(As, A, rowBase, nrows, tid);
    __syncthreads();

    const int wave = tid >> 5, lane = tid & 31;
    float* outs[4] = {C0, C1, C2, C3};
    const int widx[4] = {0, 1, 4, 3};
#pragma unroll
    for (int w = 0; w < 4; w++) {
        if (w < 3) {
            // pull the next weight matrix toward L2 while this one computes
            __builtin_prefetch(WpL + (size_t)widx[w + 1] * 65536 + (size_t)tid * 256, 0, 1);
        }
        wmma_strip(As, WpL + (size_t)widx[w] * 65536, bL + widx[w] * H,
                   outs[w], rowBase, nrows, wave, lane);
    }
}

// --------------------------- small-K embed GEMM ----------------------------
__global__ void emb_gemm(const float* __restrict__ feat, int K,
                         const float* __restrict__ W, const float* __restrict__ bias,
                         float* __restrict__ out) {
    __shared__ float srow[128];
    long row = blockIdx.x;
    int f = threadIdx.x;
    if (f < K) srow[f] = feat[row * K + f];
    __syncthreads();
    float acc = bias[f];
    for (int k = 0; k < K; k++) acc += srow[k] * W[(size_t)k * H + f];
    out[row * (long)H + f] = acc;
}

__global__ void rbf_kernel(const float* __restrict__ d, float* __restrict__ out,
                           long n, int bins, float vmin, float step, float gamma) {
    long idx = (long)blockIdx.x * blockDim.x + threadIdx.x;
    long s = (long)gridDim.x * blockDim.x;
    long total = n * bins;
    for (; idx < total; idx += s) {
        long e = idx / bins;
        int  k = (int)(idx - e * bins);
        float t = d[e] - (vmin + step * k);
        out[idx] = __expf(-gamma * t * t);
    }
}

// ------------------------------ edge gating --------------------------------
// pass1: sigma = silu(G0[dst] + G1[src] + GE) stored into GE; norm[dst] += sigma
__global__ void sigma_pass(const float* __restrict__ G0, const float* __restrict__ G1,
                           float* __restrict__ GE, const int* __restrict__ src,
                           const int* __restrict__ dst, float* __restrict__ norm) {
    long e = blockIdx.x;
    int  f = threadIdx.x;
    int  s = src[e], d = dst[e];
    float v = silu_f(G0[(size_t)d * H + f] + G1[(size_t)s * H + f] + GE[e * (long)H + f]);
    GE[e * (long)H + f] = v;
    atomicAdd(&norm[(size_t)d * H + f], v);
}

// pass2: aggr[dst] += sigma * U[src] / (norm[dst] + 1e-8)
__global__ void msg_pass(const float* __restrict__ GE, const float* __restrict__ U,
                         const float* __restrict__ norm, const int* __restrict__ src,
                         const int* __restrict__ dst, float* __restrict__ aggr) {
    long e = blockIdx.x;
    int  f = threadIdx.x;
    int  s = src[e], d = dst[e];
    float sg = GE[e * (long)H + f];
    float u  = U[(size_t)s * H + f];
    float nm = norm[(size_t)d * H + f] + 1e-8f;
    atomicAdd(&aggr[(size_t)d * H + f], sg * u / nm);
}

// ------------------------------ BatchNorm ----------------------------------
__global__ void bn_stats(const float* __restrict__ y, float* __restrict__ stat, int n) {
    int f = threadIdx.x;
    long r0 = (long)blockIdx.x * 64;
    float s = 0.0f, s2 = 0.0f;
    for (int r = 0; r < 64; r++) {
        long row = r0 + r;
        if (row < n) {
            float v = y[row * (long)H + f];
            s += v; s2 += v * v;
        }
    }
    atomicAdd(&stat[f], s);
    atomicAdd(&stat[H + f], s2);
}

__global__ void bn_apply(const float* __restrict__ yin, const float* __restrict__ stat,
                         float invn, const float* __restrict__ gamma,
                         const float* __restrict__ beta, const float* __restrict__ res,
                         float* __restrict__ out, long total) {
    long idx = (long)blockIdx.x * blockDim.x + threadIdx.x;
    long s = (long)gridDim.x * blockDim.x;
    for (; idx < total; idx += s) {
        int f = (int)(idx & (H - 1));
        float m  = stat[f] * invn;
        float vv = stat[H + f] * invn - m * m;
        float z  = gamma[f] * (yin[idx] - m) * rsqrtf(vv + 1e-5f) + beta[f];
        z = silu_f(z);
        out[idx] = res ? (z + res[idx]) : z;
    }
}

// ------------------------------ pool + head --------------------------------
__global__ void pool_kernel(const float* __restrict__ x, const int* __restrict__ batch,
                            float* __restrict__ pooled, float* __restrict__ cnt) {
    long i = blockIdx.x;
    int  f = threadIdx.x;
    int  b = batch[i];
    atomicAdd(&pooled[(size_t)b * H + f], x[i * (long)H + f]);
    if (f == 0) atomicAdd(&cnt[b], 1.0f);
}

__global__ void out_kernel(const float* __restrict__ pooled, const float* __restrict__ cnt,
                           const float* __restrict__ Wout, const float* __restrict__ bout,
                           float* __restrict__ out) {
    __shared__ float red[256];
    int b = blockIdx.x, f = threadIdx.x;
    float c = fmaxf(cnt[b], 1.0f);
    red[f] = (pooled[(size_t)b * H + f] / c) * Wout[f];
    __syncthreads();
    for (int s = 128; s > 0; s >>= 1) {
        if (f < s) red[f] += red[f + s];
        __syncthreads();
    }
    if (f == 0) out[b] = red[0] + bout[0];
}

// ------------------------------- host side ---------------------------------

static inline int gsb(long n) {   // grid-stride block count
    long b = (n + 255) / 256;
    return (int)(b > 8192 ? 8192 : (b < 1 ? 1 : b));
}

extern "C" void kernel_launch(void* const* d_in, const int* in_sizes, int n_in,
                              void* d_out, int out_size, void* d_ws, size_t ws_size,
                              hipStream_t stream) {
    (void)in_sizes; (void)n_in; (void)out_size; (void)ws_size;

    const float* x_atoms = (const float*)d_in[0];
    const float* dist    = (const float*)d_in[1];
    const float* angle   = (const float*)d_in[2];
    const int*   ei      = (const int*)d_in[3];    // [2, E]
    const int*   lg      = (const int*)d_in[4];    // [2, E3]
    const int*   batch   = (const int*)d_in[5];
    const float* W_atom  = (const float*)d_in[6];
    const float* b_atom  = (const float*)d_in[7];
    const float* bn_atom = (const float*)d_in[8];
    const float* W_edge  = (const float*)d_in[9];
    const float* b_edge  = (const float*)d_in[10];
    const float* bn_edge = (const float*)d_in[11];
    const float* W_angle = (const float*)d_in[12];
    const float* b_angle = (const float*)d_in[13];
    const float* bn_ang  = (const float*)d_in[14];
    const float* conv_W  = (const float*)d_in[15]; // [12,5,256,256]
    const float* conv_b  = (const float*)d_in[16]; // [12,5,256]
    const float* conv_bn = (const float*)d_in[17]; // [12,2,256]
    const float* W_out   = (const float*)d_in[18];
    const float* b_out   = (const float*)d_in[19];
    float* outp = (float*)d_out;

    // ---------------- workspace carve-up ----------------
    char* ws = (char*)d_ws;
    size_t off = 0;
    auto carve = [&](size_t bytes) {
        size_t o = off;
        off += (bytes + 255) & ~(size_t)255;
        return (void*)(ws + o);
    };
    float*  xBuf  = (float*) carve((size_t)NNODE * H * 4);
    float*  eBuf  = (float*) carve((size_t)NEDGE * H * 4);
    float*  aBuf  = (float*) carve((size_t)NANG  * H * 4);
    __bf16* abf   = (__bf16*)carve((size_t)NANG  * H * 2);
    __bf16* inbf  = (__bf16*)carve((size_t)NEDGE * H * 2);
    __bf16* ebf   = (__bf16*)carve((size_t)NEDGE * H * 2);
    float*  G0    = (float*) carve((size_t)NEDGE * H * 4);
    float*  G1    = (float*) carve((size_t)NEDGE * H * 4);
    float*  Ub    = (float*) carve((size_t)NEDGE * H * 4);
    float*  Ob    = (float*) carve((size_t)NEDGE * H * 4);
    float*  GE    = (float*) carve((size_t)NANG  * H * 4);  // gate GEMM out / sigma / rbf scratch
    float*  norm  = (float*) carve((size_t)NEDGE * H * 4);
    float*  aggr  = (float*) carve((size_t)NEDGE * H * 4);
    __bf16* Wp    = (__bf16*)carve((size_t)60 * 65536 * 2);
    float*  stat  = (float*) carve(2 * H * 4);
    float*  pooled= (float*) carve((size_t)NB * H * 4);
    float*  cnt   = (float*) carve((size_t)NB * 4);

    // ---------------- pack all 60 conv weights to bf16 fragment layout ------
    {
        long tot = (long)60 * 65536;
        pack_w<<<gsb(tot), 256, 0, stream>>>(conv_W, Wp, tot);
    }

    // ---------------- embeddings ----------------
    // x = silu(bn(x_atoms @ W_atom + b))
    emb_gemm<<<NNODE, 256, 0, stream>>>(x_atoms, 92, W_atom, b_atom, xBuf);
    zero_f32<<<1, 256, 0, stream>>>(stat, 2 * H);
    bn_stats<<<(NNODE + 63) / 64, 256, 0, stream>>>(xBuf, stat, NNODE);
    bn_apply<<<gsb((long)NNODE * H), 256, 0, stream>>>(xBuf, stat, 1.0f / NNODE,
        bn_atom, bn_atom + H, nullptr, xBuf, (long)NNODE * H);

    // e = silu(bn(rbf(dist,0,10,40) @ W_edge + b))
    {
        float step = 10.0f / 39.0f, g = (39.0f / 10.0f) * (39.0f / 10.0f);
        rbf_kernel<<<gsb((long)NEDGE * 40), 256, 0, stream>>>(dist, GE, NEDGE, 40, 0.0f, step, g);
        emb_gemm<<<NEDGE, 256, 0, stream>>>(GE, 40, W_edge, b_edge, eBuf);
        zero_f32<<<1, 256, 0, stream>>>(stat, 2 * H);
        bn_stats<<<(NEDGE + 63) / 64, 256, 0, stream>>>(eBuf, stat, NEDGE);
        bn_apply<<<gsb((long)NEDGE * H), 256, 0, stream>>>(eBuf, stat, 1.0f / NEDGE,
            bn_edge, bn_edge + H, nullptr, eBuf, (long)NEDGE * H);
    }

    // a = silu(bn(rbf(angle,-1,1,20) @ W_angle + b))
    {
        float step = 2.0f / 19.0f, g = (19.0f / 2.0f) * (19.0f / 2.0f);
        rbf_kernel<<<gsb((long)NANG * 20), 256, 0, stream>>>(angle, GE, NANG, 20, -1.0f, step, g);
        emb_gemm<<<NANG, 256, 0, stream>>>(GE, 20, W_angle, b_angle, aBuf);
        zero_f32<<<1, 256, 0, stream>>>(stat, 2 * H);
        bn_stats<<<(NANG + 63) / 64, 256, 0, stream>>>(aBuf, stat, NANG);
        bn_apply<<<gsb((long)NANG * H), 256, 0, stream>>>(aBuf, stat, 1.0f / NANG,
            bn_ang, bn_ang + H, nullptr, aBuf, (long)NANG * H);
        cast_bf16<<<gsb((long)NANG * H), 256, 0, stream>>>(aBuf, abf, (long)NANG * H);
    }

    // ---------------- 12 EdgeGatedGraphConv layers ----------------
    for (int l = 0; l < 12; l++) {
        bool edgeConv = (l < 8) && ((l & 1) == 0);
        float* xin = edgeConv ? eBuf : xBuf;
        int    n   = edgeConv ? NEDGE : NNODE;
        int    ne  = edgeConv ? NANG : NEDGE;
        const int* src = edgeConv ? lg : ei;
        const int* dst = edgeConv ? (lg + NANG) : (ei + NEDGE);
        const __bf16* eatt;

        cast_bf16<<<gsb((long)n * H), 256, 0, stream>>>(xin, inbf, (long)n * H);
        if (edgeConv) {
            eatt = abf;
        } else {
            cast_bf16<<<gsb((long)NEDGE * H), 256, 0, stream>>>(eBuf, ebf, (long)NEDGE * H);
            eatt = ebf;
        }

        const __bf16* WpL = Wp + (size_t)l * 5 * 65536;
        const float*  bL  = conv_b + (size_t)l * 5 * H;
        int gN  = (n  + 15) / 16;
        int gNe = (ne + 15) / 16;
        // fused: G0 = x@W0, G1 = x@W1, U = x@W4, O = x@W3 (one A-tile stage)
        gemm_wmma4<<<gN, 128, 0, stream>>>(inbf, WpL, bL, G0, G1, Ub, Ob, n);
        // edge-gate GEMM: GE = eattr @ W2
        gemm_wmma<<<gNe, 128, 0, stream>>>(eatt, WpL + 2 * 65536, bL + 2 * H, GE, ne);

        zero_f32<<<gsb((long)n * H), 256, 0, stream>>>(norm, (long)n * H);
        zero_f32<<<gsb((long)n * H), 256, 0, stream>>>(aggr, (long)n * H);

        sigma_pass<<<ne, 256, 0, stream>>>(G0, G1, GE, src, dst, norm);
        msg_pass<<<ne, 256, 0, stream>>>(GE, Ub, norm, src, dst, aggr);

        add_inplace<<<gsb((long)n * H), 256, 0, stream>>>(Ob, aggr, (long)n * H);

        zero_f32<<<1, 256, 0, stream>>>(stat, 2 * H);
        bn_stats<<<(n + 63) / 64, 256, 0, stream>>>(Ob, stat, n);
        bn_apply<<<gsb((long)n * H), 256, 0, stream>>>(Ob, stat, 1.0f / n,
            conv_bn + (size_t)l * 2 * H, conv_bn + (size_t)l * 2 * H + H,
            xin, xin, (long)n * H);
    }

    // ---------------- global mean pool + linear head ----------------
    zero_f32<<<gsb((long)NB * H), 256, 0, stream>>>(pooled, (long)NB * H);
    zero_f32<<<1, 64, 0, stream>>>(cnt, NB);
    pool_kernel<<<NNODE, 256, 0, stream>>>(xBuf, batch, pooled, cnt);
    out_kernel<<<NB, 256, 0, stream>>>(pooled, cnt, W_out, b_out, outp);
}
